// AppPreUserPGtrDocAttn_26740466385157
// MI455X (gfx1250) — compile-verified, matrix operand order checked by reading, not run on hardware
//
#include <hip/hip_runtime.h>
#include <hip/hip_bf16.h>
#include <math.h>

// ---------------------------------------------------------------------------
// Shapes
// ---------------------------------------------------------------------------
#define S_ROWS 4096     // sequence length (padded M for all GEMMs)
#define N_ROWS 4093     // valid output rows
#define APP_K  10000    // app feature dim (K of GEMM1)
#define APP_KP 10016    // K padded to multiple of 32
#define EMB    256      // app/uid embedding dim
#define TIMEB  64       // time embedding dim
#define FEAT   320      // EMB + TIMEB
#define DEC_N  10000    // decoder output dim (multiple of 16)

typedef __attribute__((ext_vector_type(16))) __bf16 v16bf;
typedef __attribute__((ext_vector_type(8)))  __bf16 v8bf;
typedef __attribute__((ext_vector_type(4)))  __bf16 v4bf;
typedef __attribute__((ext_vector_type(8)))  float  v8f;
typedef __attribute__((ext_vector_type(4)))  float  v4f;

union BfFrag { v16bf v; v8bf h[2]; };

// A/B 16x32 bf16 fragment: per lane, low 8 elems = [kb, kb+8), high 8 = [kb+16, kb+24)
__device__ __forceinline__ v16bf load_frag_bf16(const __bf16* __restrict__ base) {
  BfFrag f;
  f.h[0] = *(const v8bf*)(base);
  f.h[1] = *(const v8bf*)(base + 16);
  return f.v;
}

#define WMMA_BF16(A, B, C) \
  __builtin_amdgcn_wmma_f32_16x16x32_bf16(false, (A), false, (B), (short)0, (C), false, false)

// ---------------------------------------------------------------------------
// Prep kernels
// ---------------------------------------------------------------------------
// dst[c][r] = src[r][c] as bf16, zero pad r >= rows_src.  dst: cols_src x ld_dst
__global__ void k_cvt_transpose_pad(const float* __restrict__ src, __bf16* __restrict__ dst,
                                    int rows_src, int cols_src, int ld_dst) {
  long i = (long)blockIdx.x * blockDim.x + threadIdx.x;
  long total = (long)cols_src * ld_dst;
  if (i >= total) return;
  int c = (int)(i / ld_dst);
  int r = (int)(i % ld_dst);
  float v = (r < rows_src) ? src[(size_t)r * cols_src + c] : 0.0f;
  dst[i] = (__bf16)v;
}

__global__ void k_cvt_bf16(const float* __restrict__ src, __bf16* __restrict__ dst, long n) {
  long i = (long)blockIdx.x * blockDim.x + threadIdx.x;
  if (i < n) dst[i] = (__bf16)src[i];
}

// x[i][256 + t] = emb_tim_w[tim[i]][t]
__global__ void k_fill_x_tim(const int* __restrict__ tim, const float* __restrict__ emb_tim_w,
                             float* __restrict__ x) {
  int i = blockIdx.x * blockDim.x + threadIdx.x;
  if (i >= S_ROWS * TIMEB) return;
  int row = i >> 6, t = i & (TIMEB - 1);
  x[(size_t)row * FEAT + EMB + t] = emb_tim_w[(size_t)tim[row] * TIMEB + t];
}

// ---------------------------------------------------------------------------
// GEMM1: x[:, :256] = app(4096 x 10000, f32) @ emb_app_w via Bt(256 x 10016, bf16)
// Cooperative workgroup: 128 threads = 4 waves cover one 16-row M tile and all
// 256 output columns. The f32 A tile is converted to bf16 once per workgroup
// into a DOUBLE-BUFFERED LDS tile (80-byte padded row stride -> conflict-free
// ds_load_b128), so each K slab costs a single __syncthreads and the next
// slab's global load overlaps the current slab's 4 WMMAs.
// ---------------------------------------------------------------------------
#define LDS_STRIDE 40   // bf16 elements per LDS row: 32 data + 8 pad (80 bytes)

__global__ __launch_bounds__(128) void k_gemm_app(const float* __restrict__ A,
                                                  const __bf16* __restrict__ Bt,
                                                  float* __restrict__ X) {
  __shared__ __bf16 atile[2][16 * LDS_STRIDE];

  const int tid  = threadIdx.x;
  const int lane = tid & 31;
  const int wave = tid >> 5;      // 0..3 -> 64 output columns each
  const int rid  = lane & 15;
  const int sel  = lane >> 4;
  const int m0   = blockIdx.x * 16;
  const int n0   = wave * 64;

  // cooperative A staging: thread -> (row = tid/8, 4 floats at col (tid%8)*4)
  const int lrow = tid >> 3;
  const int lcol = (tid & 7) * 4;
  const float* Aload = A + (size_t)(m0 + lrow) * APP_K + lcol;
  __bf16* lst0 = &atile[0][lrow * LDS_STRIDE + lcol];
  __bf16* lst1 = &atile[1][lrow * LDS_STRIDE + lcol];

  // fragment read pointers (per lane)
  const __bf16* afl0 = &atile[0][rid * LDS_STRIDE + sel * 8];
  const __bf16* afl1 = &atile[1][rid * LDS_STRIDE + sel * 8];
  const __bf16* B0   = Bt + (size_t)(n0 + rid) * APP_KP + sel * 8;

  v8f acc[4];
  #pragma unroll
  for (int t = 0; t < 4; ++t) acc[t] = (v8f){};

  // helper: convert 4 floats -> 4 bf16
  auto cvt4 = [](v4f f) {
    v4bf o;
    o[0] = (__bf16)f[0]; o[1] = (__bf16)f[1];
    o[2] = (__bf16)f[2]; o[3] = (__bf16)f[3];
    return o;
  };

  // prologue: stage slab k=0 into buffer 0
  *(v4bf*)lst0 = cvt4(*(const v4f*)(Aload));
  __syncthreads();

  int buf = 0;
  int k = 0;
  // main: slabs [0, 9952) — next slab (k+32 <= 9952) is always fully in-bounds
  for (; k < 9952; k += 32) {
    __builtin_prefetch(Aload + k + 128, 0, 1);      // global_prefetch_b8 4 slabs ahead
    v4f fn = *(const v4f*)(Aload + k + 32);         // next slab, overlaps WMMAs below

    const __bf16* acur = buf ? afl1 : afl0;
    BfFrag af;
    af.h[0] = *(const v8bf*)(acur);                 // ds_load_b128
    af.h[1] = *(const v8bf*)(acur + 16);            // ds_load_b128
    #pragma unroll
    for (int t = 0; t < 4; ++t) {
      v16bf b = load_frag_bf16(B0 + (size_t)(t * 16) * APP_KP + k);
      acc[t] = WMMA_BF16(af.v, b, acc[t]);
    }

    *(v4bf*)(buf ? lst0 : lst1) = cvt4(fn);         // stage into the other buffer
    __syncthreads();                                // one barrier per slab
    buf ^= 1;
  }
  { // k == 9952: compute it, stage the guarded tail slab [9984, 10016)
    v4f fn = {};
    if (lcol < 16) fn = *(const v4f*)(Aload + 9984); // cols 9984+lcol .. +4 <= 10000
    const __bf16* acur = buf ? afl1 : afl0;
    BfFrag af;
    af.h[0] = *(const v8bf*)(acur);
    af.h[1] = *(const v8bf*)(acur + 16);
    #pragma unroll
    for (int t = 0; t < 4; ++t) {
      v16bf b = load_frag_bf16(B0 + (size_t)(t * 16) * APP_KP + 9952);
      acc[t] = WMMA_BF16(af.v, b, acc[t]);
    }
    *(v4bf*)(buf ? lst0 : lst1) = cvt4(fn);
    __syncthreads();
    buf ^= 1;
  }
  { // k == 9984: tail slab compute only (Bt zero-padded beyond K=10000)
    const __bf16* acur = buf ? afl1 : afl0;
    BfFrag af;
    af.h[0] = *(const v8bf*)(acur);
    af.h[1] = *(const v8bf*)(acur + 16);
    #pragma unroll
    for (int t = 0; t < 4; ++t) {
      v16bf b = load_frag_bf16(B0 + (size_t)(t * 16) * APP_KP + 9984);
      acc[t] = WMMA_BF16(af.v, b, acc[t]);
    }
  }

  #pragma unroll
  for (int r = 0; r < 8; ++r) {
    int row = m0 + r + sel * 8;
    #pragma unroll
    for (int t = 0; t < 4; ++t)
      X[(size_t)row * FEAT + n0 + t * 16 + rid] = acc[t][r];
  }
}

// ---------------------------------------------------------------------------
// Attention: per n, H = tanh(win @ attn_W + b), L1-normalize, weighted pool.
// Writes pooled (bf16, padded rows zeroed) and Z[:, 256:320] = ptim embedding.
// ---------------------------------------------------------------------------
__global__ void k_attn_pool(const float* __restrict__ x, const float* __restrict__ attn_W,
                            const float* __restrict__ attn_b, const int* __restrict__ ptim,
                            const float* __restrict__ emb_tim_w,
                            __bf16* __restrict__ pooled, __bf16* __restrict__ Z) {
  int n = blockIdx.x * blockDim.x + threadIdx.x;
  if (n >= S_ROWS) return;
  if (n >= N_ROWS) { // zero padding rows so the decode GEMM sees clean operands
    for (int d = 0; d < FEAT; ++d)  pooled[(size_t)n * FEAT + d] = (__bf16)0.0f;
    for (int t = 0; t < TIMEB; ++t) Z[(size_t)n * FEAT + EMB + t] = (__bf16)0.0f;
    return;
  }
  float h[4];
  float l1 = 0.0f;
  #pragma unroll
  for (int j = 0; j < 4; ++j) {
    const float* row = x + (size_t)(n + j) * FEAT;
    float acc = 0.0f;
    for (int d = 0; d < FEAT; ++d) acc += row[d] * attn_W[d];
    float t = tanhf(acc + attn_b[j]);
    h[j] = t;
    l1 += fabsf(t);
  }
  float inv = 1.0f / l1;
  #pragma unroll
  for (int j = 0; j < 4; ++j) h[j] *= inv;
  for (int d = 0; d < FEAT; ++d) {
    float p = 0.0f;
    #pragma unroll
    for (int j = 0; j < 4; ++j) p += h[j] * x[(size_t)(n + j) * FEAT + d];
    pooled[(size_t)n * FEAT + d] = (__bf16)p;
  }
  int pt = ptim[n];
  for (int t = 0; t < TIMEB; ++t)
    Z[(size_t)n * FEAT + EMB + t] = (__bf16)emb_tim_w[(size_t)pt * TIMEB + t];
}

// ---------------------------------------------------------------------------
// FC GEMM: Z[:, :256] = bf16( (pooled @ fcT + fc_b) * uid_emb )
// pooled: 4096x320 bf16, fcT: 256x320 bf16 (attn_fc_w already N x K)
// ---------------------------------------------------------------------------
__global__ __launch_bounds__(32) void k_gemm_fc(const __bf16* __restrict__ P,
                                                const __bf16* __restrict__ Wt,
                                                const float* __restrict__ fc_b,
                                                const float* __restrict__ emb_uid_w,
                                                const int* __restrict__ uid,
                                                __bf16* __restrict__ Z) {
  const int lane = threadIdx.x & 31;
  const int rid  = lane & 15;
  const int sel  = lane >> 4;
  const int m0   = blockIdx.x * 16;
  const int n0   = blockIdx.y * 32;

  const __bf16* Abase = P  + (size_t)(m0 + rid) * FEAT + sel * 8;
  const __bf16* B0    = Wt + (size_t)(n0 + rid) * FEAT + sel * 8;
  const __bf16* B1    = B0 + (size_t)16 * FEAT;

  v8f acc0 = {};
  v8f acc1 = {};
  #pragma unroll
  for (int k = 0; k < FEAT; k += 32) {
    v16bf a  = load_frag_bf16(Abase + k);
    v16bf b0 = load_frag_bf16(B0 + k);
    v16bf b1 = load_frag_bf16(B1 + k);
    acc0 = WMMA_BF16(a, b0, acc0);
    acc1 = WMMA_BF16(a, b1, acc1);
  }

  const float* urow = emb_uid_w + (size_t)uid[0] * EMB;
  #pragma unroll
  for (int r = 0; r < 8; ++r) {
    int row = m0 + r + sel * 8;
    int c0 = n0 + rid, c1 = c0 + 16;
    Z[(size_t)row * FEAT + c0] = (__bf16)((acc0[r] + fc_b[c0]) * urow[c0]);
    Z[(size_t)row * FEAT + c1] = (__bf16)((acc1[r] + fc_b[c1]) * urow[c1]);
  }
}

// ---------------------------------------------------------------------------
// Decode GEMM: out = sigmoid(Z @ dec_w.T + dec_b), Z: 4096x320 bf16,
// dec: 10000x320 bf16 (already N x K). One wave -> 16x80 tile (5 accumulators).
// ---------------------------------------------------------------------------
__global__ __launch_bounds__(32) void k_gemm_dec(const __bf16* __restrict__ Zb,
                                                 const __bf16* __restrict__ Dec,
                                                 const float* __restrict__ dec_b,
                                                 float* __restrict__ out) {
  const int lane = threadIdx.x & 31;
  const int rid  = lane & 15;
  const int sel  = lane >> 4;
  const int m0   = blockIdx.x * 16;
  const int n0   = blockIdx.y * 80;

  const __bf16* Abase = Zb + (size_t)(m0 + rid) * FEAT + sel * 8;
  const __bf16* Bbase = Dec + (size_t)(n0 + rid) * FEAT + sel * 8;

  v8f acc[5];
  #pragma unroll
  for (int t = 0; t < 5; ++t) acc[t] = (v8f){};

  #pragma unroll 2
  for (int k = 0; k < FEAT; k += 32) {
    v16bf a = load_frag_bf16(Abase + k);
    #pragma unroll
    for (int t = 0; t < 5; ++t) {
      v16bf b = load_frag_bf16(Bbase + (size_t)(t * 16) * FEAT + k);
      acc[t] = WMMA_BF16(a, b, acc[t]);
    }
  }

  #pragma unroll
  for (int t = 0; t < 5; ++t) {
    #pragma unroll
    for (int r = 0; r < 8; ++r) {
      int row = m0 + r + sel * 8;
      if (row < N_ROWS) {
        int col = n0 + t * 16 + rid;
        float v = acc[t][r] + dec_b[col];
        out[(size_t)row * DEC_N + col] = 1.0f / (1.0f + expf(-v));
      }
    }
  }
}

// ---------------------------------------------------------------------------
// Launch
// ---------------------------------------------------------------------------
extern "C" void kernel_launch(void* const* d_in, const int* in_sizes, int n_in,
                              void* d_out, int out_size, void* d_ws, size_t ws_size,
                              hipStream_t stream) {
  const int*   tim       = (const int*)  d_in[0];
  const float* app       = (const float*)d_in[1];
  // d_in[2] = loc (unused by reference)
  const int*   uid       = (const int*)  d_in[3];
  const int*   ptim      = (const int*)  d_in[4];
  const float* emb_tim_w = (const float*)d_in[5];
  const float* emb_uid_w = (const float*)d_in[6];
  const float* emb_app_w = (const float*)d_in[7];
  const float* attn_W    = (const float*)d_in[8];
  const float* attn_b    = (const float*)d_in[9];
  const float* attn_fc_w = (const float*)d_in[10];
  const float* attn_fc_b = (const float*)d_in[11];
  const float* dec_w     = (const float*)d_in[12];
  const float* dec_b     = (const float*)d_in[13];
  float* out = (float*)d_out;

  // Workspace carve-up (all 256B aligned), total 22,177,792 bytes
  char* ws = (char*)d_ws;
  __bf16* Bt     = (__bf16*)(ws + 0);          // 256 x 10016 bf16 = 5,128,192
  __bf16* decb   = (__bf16*)(ws + 5128192);    // 10000 x 320 bf16 = 6,400,000
  __bf16* fcT    = (__bf16*)(ws + 11528192);   // 256 x 320 bf16   =   163,840
  float*  X      = (float*) (ws + 11692032);   // 4096 x 320 f32   = 5,242,880
  __bf16* pooled = (__bf16*)(ws + 16934912);   // 4096 x 320 bf16  = 2,621,440
  __bf16* Zb     = (__bf16*)(ws + 19556352);   // 4096 x 320 bf16  = 2,621,440

  // 1) emb_app_w^T -> bf16, K zero-padded to 10016
  {
    long total = (long)EMB * APP_KP;
    k_cvt_transpose_pad<<<(unsigned)((total + 255) / 256), 256, 0, stream>>>(
        emb_app_w, Bt, APP_K, EMB, APP_KP);
  }
  // 2) dec_w -> bf16 (already N x K)
  {
    long total = (long)DEC_N * FEAT;
    k_cvt_bf16<<<(unsigned)((total + 255) / 256), 256, 0, stream>>>(dec_w, decb, total);
  }
  // 3) attn_fc_w -> bf16 (already N x K)
  {
    long total = (long)EMB * FEAT;
    k_cvt_bf16<<<(unsigned)((total + 255) / 256), 256, 0, stream>>>(attn_fc_w, fcT, total);
  }
  // 4) time-embedding gather into x[:, 256:320]
  k_fill_x_tim<<<(S_ROWS * TIMEB) / 256, 256, 0, stream>>>(tim, emb_tim_w, X);

  // 5) GEMM1: x[:, :256] = app @ emb_app_w (double-buffered LDS A staging)
  k_gemm_app<<<S_ROWS / 16, 128, 0, stream>>>(app, Bt, X);

  // 6) attention pooling -> pooled bf16; Z[:, 256:320] = ptim embedding
  k_attn_pool<<<S_ROWS / 128, 128, 0, stream>>>(X, attn_W, attn_b, ptim, emb_tim_w,
                                                pooled, Zb);

  // 7) FC GEMM: Z[:, :256] = (pooled @ fc.T + fc_b) * uid_emb
  k_gemm_fc<<<dim3(S_ROWS / 16, EMB / 32), 32, 0, stream>>>(pooled, fcT, attn_fc_b,
                                                            emb_uid_w, uid, Zb);

  // 8) decode GEMM + sigmoid -> d_out (N_ROWS x 10000 f32)
  k_gemm_dec<<<dim3(S_ROWS / 16, DEC_N / 80), 32, 0, stream>>>(Zb, decb, dec_b, out);
}
